// MultiHeadAttention_43155831390829
// MI455X (gfx1250) — matrix operand order repaired and checked by reading, not compiled, not run on hardware
//
#include <hip/hip_runtime.h>

// ---------------------------------------------------------------------------
// MultiHeadAttention for MI455X (gfx1250, wave32, WMMA)
//   HIDDEN=1024, HEADS=16, HEAD_DIM=64, BSZ=2, SEQ=2048, causal
// Pipeline:
//   1) gemm<float,QKV>   : X(f32) x W(f32) -> Q/K/V bf16 in [B,H,S,D] (Q pre-scaled)
//   2) attn              : flash-attention with v_wmma_f32_16x16x32_bf16,
//                          DPP row reductions, async LDS staging when available
//   3) gemm<bf16,PLAIN>  : O_attn(bf16) x Wo(f32) -> out f32
// Workspace: 4 x 4194304 bf16 = 32 MB
// ---------------------------------------------------------------------------

typedef __attribute__((ext_vector_type(16))) __bf16 bf16x16;
typedef __attribute__((ext_vector_type(2)))  __bf16 bf16x2;
typedef __attribute__((ext_vector_type(8)))  float  floatx8;
typedef __attribute__((ext_vector_type(4)))  int    i32x4;
typedef i32x4 __attribute__((address_space(1))) gi32x4;   // global
typedef i32x4 __attribute__((address_space(3))) li32x4;   // LDS

#define HIDDEN   1024
#define HEADS    16
#define HEAD_DIM 64
#define BSZ      2
#define SEQ      2048
#define INNER    (HEADS * HEAD_DIM)
#define SCALE_F  0.125f  // 64^-0.5

#if __has_builtin(__builtin_amdgcn_global_load_async_to_lds_b128)
#define HAVE_ASYNC_LDS 1
#else
#define HAVE_ASYNC_LDS 0
#endif

__device__ inline void wait_async0() {
#if HAVE_ASYNC_LDS
#if __has_builtin(__builtin_amdgcn_s_wait_asynccnt)
  __builtin_amdgcn_s_wait_asynccnt(0);
#else
  asm volatile("s_wait_asynccnt 0" ::: "memory");
#endif
#endif
}

// Native f32 -> bf16 (RNE); backend lowers fptrunc-to-bf16 with hw cvt ops.
__device__ inline __bf16 f2bf(float f) { return static_cast<__bf16>(f); }

// Packed f32x2 -> bf16x2 (native v_cvt_pk_bf16_f32 when available)
__device__ inline bf16x2 f2bf2(float a, float b) {
#if __has_builtin(__builtin_amdgcn_cvt_pk_bf16_f32)
  auto r = __builtin_amdgcn_cvt_pk_bf16_f32(a, b);
  return __builtin_bit_cast(bf16x2, r);
#else
  bf16x2 v; v[0] = f2bf(a); v[1] = f2bf(b); return v;
#endif
}

union FragU { bf16x16 v; uint4 q[2]; };
union Pack4 { __bf16 h[4]; bf16x2 h2[2]; uint2 u; };
union Pack8 { uint4 u; __bf16 h[8]; };

// ---- DPP 16-lane row reductions (halves of the wave stay independent) ----
template <int CTRL>
__device__ inline float dpp_movf(float x) {
  int i = __builtin_bit_cast(int, x);
  int r = __builtin_amdgcn_update_dpp(i, i, CTRL, 0xF, 0xF, false);
  return __builtin_bit_cast(float, r);
}
__device__ inline float row_max16(float x) {
  x = fmaxf(x, dpp_movf<0xB1>(x));    // quad_perm [1,0,3,2]  (xor 1)
  x = fmaxf(x, dpp_movf<0x4E>(x));    // quad_perm [2,3,0,1]  (xor 2)
  x = fmaxf(x, dpp_movf<0x124>(x));   // row_ror:4
  x = fmaxf(x, dpp_movf<0x128>(x));   // row_ror:8
  return x;
}
__device__ inline float row_sum16(float x) {
  x += dpp_movf<0xB1>(x);
  x += dpp_movf<0x4E>(x);
  x += dpp_movf<0x124>(x);
  x += dpp_movf<0x128>(x);
  return x;
}

// A-matrix 16x32 bf16 fragment (ISA 7.12.2): lane&15 = row M,
// lanes 0-15 hold K = koff + {0..7, 16..23}, lanes 16-31 hold K = koff + {8..15, 24..31}
__device__ inline bf16x16 ld_frag_mrow(const __bf16* base, int stride, int koff, int lane) {
  int row = lane & 15;
  int kb  = (lane >> 4) << 3;
  const __bf16* p = base + row * stride + koff + kb;
  FragU f;
  f.q[0] = *(const uint4*)(p);
  f.q[1] = *(const uint4*)(p + 16);
  return f.v;
}

// B-matrix 32x16 bf16 fragment from K-contiguous ("transposed") storage [n][k]:
// lane&15 = column N, lanes 0-15 hold K = koff + {0..7, 16..23}, lanes 16-31 hold K = koff + {8..15, 24..31}
__device__ inline bf16x16 ld_frag_ncol(const __bf16* base, int stride, int koff, int lane) {
  int col = lane & 15;
  int kb  = (lane >> 4) << 3;
  const __bf16* p = base + col * stride + koff + kb;
  FragU f;
  f.q[0] = *(const uint4*)(p);
  f.q[1] = *(const uint4*)(p + 16);
  return f.v;
}

__device__ inline floatx8 wmma_bf16(bf16x16 a, bf16x16 b, floatx8 c) {
  return __builtin_amdgcn_wmma_f32_16x16x32_bf16(false, a, false, b, (short)0, c, false, false);
}

// ---------------------------------------------------------------------------
// GEMM: C[M,N] = A[M,K] * B[K,N]; A f32 or bf16, B f32 (converted to bf16).
// OUT_MODE 0: f32 row-major.  1: bf16 -> [B,H,S,D].  2: like 1, scaled by SCALE_F.
// Block 256 thr (8 waves), tile 128x128x32; wave computes 32x64 (2x4 tiles).
// ---------------------------------------------------------------------------
template <typename AT, int OUT_MODE>
__global__ __launch_bounds__(256) void gemm_kernel(const AT* __restrict__ A,
                                                   const float* __restrict__ Bw,
                                                   void* __restrict__ Cp,
                                                   int M, int N, int Kd) {
  __shared__ __bf16 As[128 * 40];   // [m][k] stride 40 (16B-aligned rows)
  __shared__ __bf16 Bs[128 * 40];   // [n][k] transposed, stride 40
  const int tid  = threadIdx.x;
  const int lane = tid & 31;
  const int wave = tid >> 5;
  const int wm   = (wave >> 1) * 32;
  const int wn   = (wave & 1) * 64;
  const int mb   = blockIdx.y * 128;
  const int nb   = blockIdx.x * 128;

  floatx8 acc[2][4] = {};

  for (int kb = 0; kb < Kd; kb += 32) {
    __syncthreads();
    // ---- stage A tile (convert f32 -> bf16 if needed) ----
    if constexpr (sizeof(AT) == 4) {
      #pragma unroll
      for (int it = 0; it < 4; ++it) {
        int row = (tid >> 3) + it * 32;
        int col = (tid & 7) * 4;
        float4 v = *(const float4*)((const float*)A + (size_t)(mb + row) * Kd + kb + col);
        Pack4 p;
        p.h2[0] = f2bf2(v.x, v.y);
        p.h2[1] = f2bf2(v.z, v.w);
        *(uint2*)(As + row * 40 + col) = p.u;
      }
    } else {
      #pragma unroll
      for (int it = 0; it < 2; ++it) {
        int row = (tid >> 2) + it * 64;
        int col = (tid & 3) * 8;
        *(uint4*)(As + row * 40 + col) =
            *(const uint4*)((const __bf16*)A + (size_t)(mb + row) * Kd + kb + col);
      }
    }
    // ---- stage B tile transposed: Bs[n][k] = Bw[kb+k][nb+n] ----
    // 4 k-strided b32 loads per thread, one packed 8-byte LDS store.
    #pragma unroll
    for (int it = 0; it < 4; ++it) {
      int n  = (tid & 127);
      int kq = (tid >> 7) * 4 + it * 8;
      const float* bp = Bw + (size_t)(kb + kq) * N + nb + n;
      Pack4 p;
      p.h2[0] = f2bf2(bp[0], bp[(size_t)N]);
      p.h2[1] = f2bf2(bp[(size_t)2 * N], bp[(size_t)3 * N]);
      *(uint2*)(Bs + n * 40 + kq) = p.u;
    }
    __syncthreads();

    bf16x16 af[2], bf[4];
    #pragma unroll
    for (int i = 0; i < 2; ++i) af[i] = ld_frag_mrow(As + (wm + i * 16) * 40, 40, 0, lane);
    #pragma unroll
    for (int n = 0; n < 4; ++n) bf[n] = ld_frag_ncol(Bs + (wn + n * 16) * 40, 40, 0, lane);
    #pragma unroll
    for (int i = 0; i < 2; ++i)
      #pragma unroll
      for (int n = 0; n < 4; ++n)
        acc[i][n] = wmma_bf16(af[i], bf[n], acc[i][n]);
  }

  // ---- epilogue: C layout lane&15 = N, VGPR r -> M = r (+8 for lanes 16-31) ----
  const int colL = lane & 15;
  const int rb   = (lane & 16) ? 8 : 0;
  #pragma unroll
  for (int i = 0; i < 2; ++i) {
    if constexpr (OUT_MODE == 0) {
      #pragma unroll
      for (int n = 0; n < 4; ++n) {
        int col = nb + wn + n * 16 + colL;
        #pragma unroll
        for (int r = 0; r < 8; ++r) {
          int row = mb + wm + i * 16 + rb + r;
          ((float*)Cp)[(size_t)row * N + col] = acc[i][n][r];
        }
      }
    } else {
      // pack pairs of n-tiles (cols 16 apart) through packed cvt
      #pragma unroll
      for (int np = 0; np < 2; ++np) {
        int n0 = 2 * np;
        int col0 = nb + wn + n0 * 16 + colL;
        #pragma unroll
        for (int r = 0; r < 8; ++r) {
          int row = mb + wm + i * 16 + rb + r;
          float v0 = acc[i][n0][r], v1 = acc[i][n0 + 1][r];
          if constexpr (OUT_MODE == 2) { v0 *= SCALE_F; v1 *= SCALE_F; }
          bf16x2 pk = f2bf2(v0, v1);
          int b = row >> 11, s = row & (SEQ - 1);
          int h0 = col0 >> 6, d0 = col0 & (HEAD_DIM - 1);
          __bf16* dst = (__bf16*)Cp + (((size_t)(b * HEADS + h0)) * SEQ + s) * HEAD_DIM + d0;
          dst[0]  = pk[0];
          dst[16] = pk[1];   // same head: d0+16 < 64 for these tile pairs
        }
      }
    }
  }
}

// ---------------------------------------------------------------------------
// Flash attention: grid (SEQ/128, BSZ*HEADS), block 256 (8 waves x 16 q-rows).
// K-blocks of 64 keys; online softmax; WMMA for S=Q*K^T and O=P*V.
// Q arrives pre-scaled by SCALE_F.
// ---------------------------------------------------------------------------
__global__ __launch_bounds__(256) void attn_kernel(const __bf16* __restrict__ Q,
                                                   const __bf16* __restrict__ K,
                                                   const __bf16* __restrict__ V,
                                                   __bf16* __restrict__ Oattn) {
  __shared__ __bf16 Ks[64 * 72];       // K block [s][d], stride 72
  __shared__ __bf16 Vt[64 * 72];       // V block transposed [d][s], stride 72
  __shared__ __bf16 Ps[8 * 16 * 72];   // per-wave P tile 16x64, stride 72

  const int tid  = threadIdx.x;
  const int lane = tid & 31;
  const int wave = tid >> 5;
  const int qb   = blockIdx.x * 128;
  const int bh   = blockIdx.y;                    // b*HEADS + h
  const int b    = bh >> 4, h = bh & 15;
  const size_t hoff = (size_t)bh * SEQ * HEAD_DIM;

  // Q fragments for this wave's 16 rows (A layout, straight from global)
  const __bf16* qbase = Q + hoff + (size_t)(qb + wave * 16) * HEAD_DIM;
  bf16x16 qf[2];
  #pragma unroll
  for (int c = 0; c < 2; ++c) qf[c] = ld_frag_mrow(qbase, HEAD_DIM, c * 32, lane);

  float m_i[8], l_i[8];
  floatx8 o[4] = {};
  #pragma unroll
  for (int r = 0; r < 8; ++r) { m_i[r] = -1e30f; l_i[r] = 0.f; }

  const int colL = lane & 15;
  const int rb   = (lane & 16) ? 8 : 0;
  const int jmax = (qb + 127) >> 6;

  for (int j = 0; j <= jmax; ++j) {
    const int jb = j * 64;
    __syncthreads();
    // ---- cooperative stage of K (row-major, async to LDS) and V (transposed) ----
    #pragma unroll
    for (int it = 0; it < 2; ++it) {
      int row  = (tid >> 3) + it * 32;
      int col8 = (tid & 7) << 3;
      const __bf16* kp = K + hoff + (size_t)(jb + row) * HEAD_DIM + col8;
#if HAVE_ASYNC_LDS
      __builtin_amdgcn_global_load_async_to_lds_b128(
          (gi32x4*)kp, (li32x4*)(Ks + row * 72 + col8), 0, 0);
#else
      *(uint4*)(Ks + row * 72 + col8) = *(const uint4*)kp;
#endif
      Pack8 vv;
      vv.u = *(const uint4*)(V + hoff + (size_t)(jb + row) * HEAD_DIM + col8);
      #pragma unroll
      for (int e = 0; e < 8; ++e) Vt[(col8 + e) * 72 + row] = vv.h[e];
    }
    if (j < jmax) {  // prefetch next K/V block -> global_prefetch_b8
      int row  = (tid >> 3);
      int col8 = (tid & 7) << 3;
      __builtin_prefetch(K + hoff + (size_t)(jb + 64 + row) * HEAD_DIM + col8, 0, 1);
      __builtin_prefetch(V + hoff + (size_t)(jb + 64 + row) * HEAD_DIM + col8, 0, 1);
    }
    wait_async0();
    __syncthreads();

    // ---- S = Q * K^T  (4 n-tiles of 16 keys, K-dim = HEAD_DIM in 2 chunks) ----
    floatx8 sc[4] = {};
    #pragma unroll
    for (int nt = 0; nt < 4; ++nt)
      #pragma unroll
      for (int c = 0; c < 2; ++c) {
        bf16x16 kf = ld_frag_ncol(Ks + nt * 16 * 72, 72, c * 32, lane);
        sc[nt] = wmma_bf16(qf[c], kf, sc[nt]);
      }

    // ---- causal mask only on diagonal blocks (wave-uniform scalar branch) ----
    const bool needMask = (jb + 63) > (qb + wave * 16);
    if (needMask) {
      #pragma unroll
      for (int r = 0; r < 8; ++r) {
        int rowg = qb + wave * 16 + rb + r;
        #pragma unroll
        for (int nt = 0; nt < 4; ++nt) {
          int colg = jb + nt * 16 + colL;
          sc[nt][r] += (colg <= rowg ? 0.f : -10000.f);
        }
      }
    }

    // ---- online softmax (rows striped: VGPR r -> row rb+r, cols across 16 lanes) ----
    float alpha[8];
    #pragma unroll
    for (int r = 0; r < 8; ++r) {
      float mx = fmaxf(fmaxf(sc[0][r], sc[1][r]), fmaxf(sc[2][r], sc[3][r]));
      mx = row_max16(mx);
      float mnew = fmaxf(m_i[r], mx);
      alpha[r] = __expf(m_i[r] - mnew);
      m_i[r] = mnew;
      float rs = 0.f;
      #pragma unroll
      for (int nt = 0; nt < 4; ++nt) {
        float p = __expf(sc[nt][r] - mnew);
        sc[nt][r] = p;
        rs += p;
      }
      rs = row_sum16(rs);
      l_i[r] = l_i[r] * alpha[r] + rs;
    }
    #pragma unroll
    for (int nt = 0; nt < 4; ++nt)
      #pragma unroll
      for (int r = 0; r < 8; ++r) o[nt][r] *= alpha[r];

    // ---- P: C-layout -> LDS (packed cvt) -> A-layout fragments ----
    __bf16* pw = Ps + wave * 16 * 72;
    #pragma unroll
    for (int np = 0; np < 2; ++np) {
      int n0 = 2 * np;
      #pragma unroll
      for (int r = 0; r < 8; ++r) {
        bf16x2 pk = f2bf2(sc[n0][r], sc[n0 + 1][r]);
        __bf16* d = pw + (rb + r) * 72 + n0 * 16 + colL;
        d[0]  = pk[0];
        d[16] = pk[1];
      }
    }
    asm volatile("s_wait_dscnt 0" ::: "memory");

    bf16x16 pf[2];
    #pragma unroll
    for (int c = 0; c < 2; ++c) pf[c] = ld_frag_mrow(pw, 72, c * 32, lane);

    // ---- O += P * V  (B from transposed Vt: lane = head-dim col, K = key pos) ----
    #pragma unroll
    for (int nt = 0; nt < 4; ++nt)
      #pragma unroll
      for (int c = 0; c < 2; ++c) {
        bf16x16 vf = ld_frag_ncol(Vt + nt * 16 * 72, 72, c * 32, lane);
        o[nt] = wmma_bf16(pf[c], vf, o[nt]);
      }
  }

  // ---- epilogue: normalize, store bf16 to O_attn [B,S,H*D] ----
  float inv[8];
  #pragma unroll
  for (int r = 0; r < 8; ++r) inv[r] = 1.0f / l_i[r];
  #pragma unroll
  for (int np = 0; np < 2; ++np) {
    int n0 = 2 * np;
    #pragma unroll
    for (int r = 0; r < 8; ++r) {
      int s = qb + wave * 16 + rb + r;
      bf16x2 pk = f2bf2(o[n0][r] * inv[r], o[n0 + 1][r] * inv[r]);
      __bf16* d = Oattn + ((size_t)(b * SEQ + s)) * INNER + h * HEAD_DIM + n0 * 16 + colL;
      d[0]  = pk[0];
      d[16] = pk[1];
    }
  }
}

// ---------------------------------------------------------------------------
extern "C" void kernel_launch(void* const* d_in, const int* in_sizes, int n_in,
                              void* d_out, int out_size, void* d_ws, size_t ws_size,
                              hipStream_t stream) {
  const float* query = (const float*)d_in[0];
  const float* key   = (const float*)d_in[1];
  const float* value = (const float*)d_in[2];
  const float* w_q   = (const float*)d_in[3];
  const float* w_k   = (const float*)d_in[4];
  const float* w_v   = (const float*)d_in[5];
  const float* w_o   = (const float*)d_in[6];

  const int M = BSZ * SEQ;              // 4096
  const size_t SZ = (size_t)M * INNER;  // 4194304 elems per buffer

  __bf16* Qb = (__bf16*)d_ws;           // [B,H,S,D] (pre-scaled)
  __bf16* Kb = Qb + SZ;
  __bf16* Vb = Kb + SZ;
  __bf16* Ob = Vb + SZ;                 // [B,S,H*D]  (total 32 MB of ws)

  dim3 blk(256);
  dim3 gp(INNER / 128, M / 128);        // (8, 32)

  gemm_kernel<float, 2><<<gp, blk, 0, stream>>>(query, w_q, (void*)Qb, M, INNER, HIDDEN);
  gemm_kernel<float, 1><<<gp, blk, 0, stream>>>(key,   w_k, (void*)Kb, M, INNER, HIDDEN);
  gemm_kernel<float, 1><<<gp, blk, 0, stream>>>(value, w_v, (void*)Vb, M, INNER, HIDDEN);

  dim3 ga(SEQ / 128, BSZ * HEADS);      // (16, 32)
  attn_kernel<<<ga, blk, 0, stream>>>(Qb, Kb, Vb, Ob);

  gemm_kernel<__bf16, 0><<<gp, blk, 0, stream>>>(Ob, w_o, d_out, M, HIDDEN, HIDDEN);
}
